// GroupQueryAttention_55027120996459
// MI455X (gfx1250) — matrix operand order repaired and checked by reading, not compiled, not run on hardware
//
#include <hip/hip_runtime.h>
#include <math.h>

#define HIDDEN 2048
#define QH 32
#define KVH 8
#define HD 64
#define BB 2
#define SS 2048
#define KVDIM (KVH * HD)   // 512

typedef __attribute__((ext_vector_type(16))) __bf16 v16bf;
typedef __attribute__((ext_vector_type(8)))  float  v8f;
typedef __attribute__((ext_vector_type(4)))  unsigned int v4u;

union Frag {
    v16bf bf;
    v4u   q[2];
};

__device__ __forceinline__ unsigned short f2bf(float f) {
    unsigned int u = __float_as_uint(f);
    unsigned int r = 0x7FFFu + ((u >> 16) & 1u);
    return (unsigned short)((u + r) >> 16);
}

// ds_swizzle group-of-32: offset = (xor<<10)|(or<<5)|and ; SWAPX masks keep 16-lane halves closed.
#define SWZMAX(x, imm) fmaxf((x), __int_as_float(__builtin_amdgcn_ds_swizzle(__float_as_int(x), (imm))))

// ---------------- conversion kernels ----------------
__global__ __launch_bounds__(256) void cvt_bf16(const float* __restrict__ in,
                                                unsigned short* __restrict__ out, long n) {
    long i = (long)blockIdx.x * blockDim.x + threadIdx.x;
    if (i < n) out[i] = f2bf(in[i]);
}

// in: K x N (row-major fp32) -> out: N x K (row-major bf16)
__global__ __launch_bounds__(256) void transpose_bf16(const float* __restrict__ in,
                                                      unsigned short* __restrict__ out,
                                                      int K, int N) {
    long i = (long)blockIdx.x * blockDim.x + threadIdx.x;
    long total = (long)K * N;
    if (i < total) {
        int k = (int)(i / N);
        int n = (int)(i % N);
        out[(long)n * K + k] = f2bf(in[i]);
    }
}

// ---------------- WMMA GEMM, 32x64 macro-tile per wave (2x4 16x16 tiles) ----------------
// C[M,N] = (A[M,K] bf16 row-major) * (Bt[N,K] bf16 N-major)^T + bias[N], then * outScale
// MODE 0: bf16 row-major out. MODE 1: bf16 out stored as (b,kh,d,s). MODE 2: fp32 row-major out.
template <int MODE>
__global__ __launch_bounds__(128) void wmma_gemm_kernel(const unsigned short* __restrict__ A,
                                                        const unsigned short* __restrict__ Bt,
                                                        const float* __restrict__ bias,
                                                        void* __restrict__ Cout,
                                                        int M, int N, int K, float outScale) {
    int wave = blockIdx.x * (blockDim.x >> 5) + (threadIdx.x >> 5);
    int lane = threadIdx.x & 31;
    int gN = N >> 6;                      // 64-wide groups
    int groups = (M >> 5) * gN;           // 32-tall groups
    if (wave >= groups) return;
    int gm = wave / gN, gn = wave - gm * gN;
    int m0 = gm << 5, n0 = gn << 6;

    int lr = lane & 15;
    int half = lane >> 4;

    const unsigned short* aR0 = A + (long)(m0 + lr) * K;
    const unsigned short* aR1 = A + (long)(m0 + 16 + lr) * K;
    const unsigned short* bR0 = Bt + (long)(n0 + lr) * K;
    const unsigned short* bR1 = Bt + (long)(n0 + 16 + lr) * K;
    const unsigned short* bR2 = Bt + (long)(n0 + 32 + lr) * K;
    const unsigned short* bR3 = Bt + (long)(n0 + 48 + lr) * K;

    v8f acc[2][4] = {};
    for (int kk = 0; kk < K; kk += 32) {
        Frag a0, a1, b0, b1, b2, b3;
        a0.q[0] = *(const v4u*)(aR0 + kk + half * 16);
        a0.q[1] = *(const v4u*)(aR0 + kk + half * 16 + 8);
        a1.q[0] = *(const v4u*)(aR1 + kk + half * 16);
        a1.q[1] = *(const v4u*)(aR1 + kk + half * 16 + 8);
        b0.q[0] = *(const v4u*)(bR0 + kk + half * 16);
        b0.q[1] = *(const v4u*)(bR0 + kk + half * 16 + 8);
        b1.q[0] = *(const v4u*)(bR1 + kk + half * 16);
        b1.q[1] = *(const v4u*)(bR1 + kk + half * 16 + 8);
        b2.q[0] = *(const v4u*)(bR2 + kk + half * 16);
        b2.q[1] = *(const v4u*)(bR2 + kk + half * 16 + 8);
        b3.q[0] = *(const v4u*)(bR3 + kk + half * 16);
        b3.q[1] = *(const v4u*)(bR3 + kk + half * 16 + 8);
        if (kk + 256 < K) {
            __builtin_prefetch(aR0 + kk + 256, 0, 0);
            __builtin_prefetch(bR0 + kk + 256, 0, 0);
        }
        acc[0][0] = __builtin_amdgcn_wmma_f32_16x16x32_bf16(false, a0.bf, false, b0.bf, (short)0, acc[0][0], false, false);
        acc[0][1] = __builtin_amdgcn_wmma_f32_16x16x32_bf16(false, a0.bf, false, b1.bf, (short)0, acc[0][1], false, false);
        acc[0][2] = __builtin_amdgcn_wmma_f32_16x16x32_bf16(false, a0.bf, false, b2.bf, (short)0, acc[0][2], false, false);
        acc[0][3] = __builtin_amdgcn_wmma_f32_16x16x32_bf16(false, a0.bf, false, b3.bf, (short)0, acc[0][3], false, false);
        acc[1][0] = __builtin_amdgcn_wmma_f32_16x16x32_bf16(false, a1.bf, false, b0.bf, (short)0, acc[1][0], false, false);
        acc[1][1] = __builtin_amdgcn_wmma_f32_16x16x32_bf16(false, a1.bf, false, b1.bf, (short)0, acc[1][1], false, false);
        acc[1][2] = __builtin_amdgcn_wmma_f32_16x16x32_bf16(false, a1.bf, false, b2.bf, (short)0, acc[1][2], false, false);
        acc[1][3] = __builtin_amdgcn_wmma_f32_16x16x32_bf16(false, a1.bf, false, b3.bf, (short)0, acc[1][3], false, false);
    }

    for (int nt = 0; nt < 4; ++nt) {
        int col = n0 + nt * 16 + lr;
        float bv = bias ? bias[col] : 0.0f;
        for (int mt = 0; mt < 2; ++mt) {
            for (int r = 0; r < 8; ++r) {
                int m = m0 + mt * 16 + r + half * 8;
                float val = (acc[mt][nt][r] + bv) * outScale;
                if (MODE == 0) {
                    ((unsigned short*)Cout)[(long)m * N + col] = f2bf(val);
                } else if (MODE == 1) {
                    int kh = col >> 6, d = col & 63;
                    int bidx = m >> 11;            // / SS
                    int s = m & (SS - 1);
                    ((unsigned short*)Cout)[((long)(bidx * KVH + kh) * HD + d) * SS + s] = f2bf(val);
                } else {
                    ((float*)Cout)[(long)m * N + col] = val;
                }
            }
        }
    }
}

// ---------------- flash attention (GQA) ----------------
// Q: (B*S, 2048) bf16, PRE-SCALED by (1/sqrt(d))*log2(e)  ->  scores already in exp2 domain.
// Km: (B*S, 512) bf16 ; Vt: (B*KVH*HD, S) bf16 ; mask: (B,1,S,S) int ; O: (B*S, 2048) bf16
__global__ __launch_bounds__(128) void gqa_attention_kernel(const unsigned short* __restrict__ Q,
                                                            const unsigned short* __restrict__ Km,
                                                            const unsigned short* __restrict__ Vt,
                                                            const int* __restrict__ mask,
                                                            unsigned short* __restrict__ O) {
    __shared__ __align__(16) unsigned short Plds[4][16 * 32];

    int waveInBlk = threadIdx.x >> 5;
    int lane = threadIdx.x & 31;
    int wave = blockIdx.x * 4 + waveInBlk;

    const int qTiles = SS / 16;              // 128
    int qt = wave % qTiles;
    int bh = wave / qTiles;                  // 0..B*QH-1
    int h = bh % QH;
    int b = bh / QH;
    int kh = h % KVH;                        // jnp.tile head ordering

    int lr = lane & 15;
    int half = lane >> 4;
    int q0 = qt << 4;

    // Q A-fragments for d=0..31 and d=32..63 (row = lr)
    const unsigned short* qrow = Q + ((long)(b * SS + q0 + lr)) * HIDDEN + h * HD;
    Frag qa0, qa1;
    qa0.q[0] = *(const v4u*)(qrow + half * 16);
    qa0.q[1] = *(const v4u*)(qrow + half * 16 + 8);
    qa1.q[0] = *(const v4u*)(qrow + 32 + half * 16);
    qa1.q[1] = *(const v4u*)(qrow + 32 + half * 16 + 8);

    // constant all-ones bf16 B-fragment (rowsum via WMMA)
    Frag onesf;
    {
        v4u o; o[0] = 0x3F803F80u; o[1] = 0x3F803F80u; o[2] = 0x3F803F80u; o[3] = 0x3F803F80u;
        onesf.q[0] = o; onesf.q[1] = o;
    }

    v8f acc0 = {}, acc1 = {}, acc2 = {}, acc3 = {};
    v8f accl = {};                            // softmax denominator per row (replicated over lanes)
    float mrow[8];
    for (int r = 0; r < 8; ++r) mrow[r] = -INFINITY;

    const long maskBase = (long)b * SS * SS;
    unsigned short* pbuf = &Plds[waveInBlk][0];
    const unsigned short* vbase = Vt + ((long)(b * KVH + kh) * HD) * SS;

    for (int jb = 0; jb < SS; jb += 32) {
        // ---- scores: 16x32 block as two 16x16 WMMA tiles (already log2-scaled via Q)
        v8f s0 = {}, s1 = {};
        {
            int key = jb + lr;
            const unsigned short* krow = Km + ((long)(b * SS + key)) * KVDIM + kh * HD;
            Frag f0, f1;
            f0.q[0] = *(const v4u*)(krow + half * 16);
            f0.q[1] = *(const v4u*)(krow + half * 16 + 8);
            f1.q[0] = *(const v4u*)(krow + 32 + half * 16);
            f1.q[1] = *(const v4u*)(krow + 32 + half * 16 + 8);
            s0 = __builtin_amdgcn_wmma_f32_16x16x32_bf16(false, qa0.bf, false, f0.bf, (short)0, s0, false, false);
            s0 = __builtin_amdgcn_wmma_f32_16x16x32_bf16(false, qa1.bf, false, f1.bf, (short)0, s0, false, false);
        }
        {
            int key = jb + 16 + lr;
            const unsigned short* krow = Km + ((long)(b * SS + key)) * KVDIM + kh * HD;
            Frag f0, f1;
            f0.q[0] = *(const v4u*)(krow + half * 16);
            f0.q[1] = *(const v4u*)(krow + half * 16 + 8);
            f1.q[0] = *(const v4u*)(krow + 32 + half * 16);
            f1.q[1] = *(const v4u*)(krow + 32 + half * 16 + 8);
            s1 = __builtin_amdgcn_wmma_f32_16x16x32_bf16(false, qa0.bf, false, f0.bf, (short)0, s1, false, false);
            s1 = __builtin_amdgcn_wmma_f32_16x16x32_bf16(false, qa1.bf, false, f1.bf, (short)0, s1, false, false);
        }

        // ---- mask + row max (ds_swizzle butterflies stay inside 16-lane halves)
        int key0 = jb + lr;
        float mnew[8];
        for (int r = 0; r < 8; ++r) {
            int qi = q0 + r + half * 8;
            long mb = maskBase + (long)qi * SS;
            float v0 = s0[r];
            float v1 = s1[r];
            if (mask[mb + key0] == 0)      v0 = -INFINITY;
            if (mask[mb + key0 + 16] == 0) v1 = -INFINITY;
            s0[r] = v0; s1[r] = v1;
            float t = fmaxf(v0, v1);
            t = SWZMAX(t, 0x041F);   // SWAPX1
            t = SWZMAX(t, 0x081F);   // SWAPX2
            t = SWZMAX(t, 0x101F);   // SWAPX4
            t = SWZMAX(t, 0x201F);   // SWAPX8
            mnew[r] = fmaxf(mrow[r], t);
        }

        // ---- exp2, rescale accumulators, write P (bf16) to LDS
        for (int r = 0; r < 8; ++r) {
            float p0, p1, alpha;
            if (mnew[r] == -INFINITY) {
                p0 = 0.0f; p1 = 0.0f; alpha = 0.0f;
            } else {
                p0 = __builtin_amdgcn_exp2f(s0[r] - mnew[r]);
                p1 = __builtin_amdgcn_exp2f(s1[r] - mnew[r]);
                alpha = (mrow[r] == -INFINITY) ? 0.0f : __builtin_amdgcn_exp2f(mrow[r] - mnew[r]);
            }
            mrow[r] = mnew[r];
            acc0[r] *= alpha; acc1[r] *= alpha; acc2[r] *= alpha; acc3[r] *= alpha;
            accl[r] *= alpha;
            int prow = r + half * 8;
            pbuf[prow * 32 + lr]      = f2bf(p0);
            pbuf[prow * 32 + 16 + lr] = f2bf(p1);
        }
        asm volatile("s_wait_dscnt 0" ::: "memory");
        Frag pa;
        const unsigned short* prd = pbuf + lr * 32;
        pa.q[0] = *(const v4u*)(prd + half * 16);
        pa.q[1] = *(const v4u*)(prd + half * 16 + 8);

        // ---- denominator: accl += P x ones  (every column = rowsum(P))
        accl = __builtin_amdgcn_wmma_f32_16x16x32_bf16(false, pa.bf, false, onesf.bf, (short)0, accl, false, false);

        // ---- acc += P * V  (four d-tiles), V^T layout -> contiguous B-fragments
        long koff = jb + half * 16;
        Frag vf;
        const unsigned short* vp;
        vp = vbase + (long)(lr) * SS + koff;
        vf.q[0] = *(const v4u*)vp; vf.q[1] = *(const v4u*)(vp + 8);
        acc0 = __builtin_amdgcn_wmma_f32_16x16x32_bf16(false, pa.bf, false, vf.bf, (short)0, acc0, false, false);
        vp = vbase + (long)(16 + lr) * SS + koff;
        vf.q[0] = *(const v4u*)vp; vf.q[1] = *(const v4u*)(vp + 8);
        acc1 = __builtin_amdgcn_wmma_f32_16x16x32_bf16(false, pa.bf, false, vf.bf, (short)0, acc1, false, false);
        vp = vbase + (long)(32 + lr) * SS + koff;
        vf.q[0] = *(const v4u*)vp; vf.q[1] = *(const v4u*)(vp + 8);
        acc2 = __builtin_amdgcn_wmma_f32_16x16x32_bf16(false, pa.bf, false, vf.bf, (short)0, acc2, false, false);
        vp = vbase + (long)(48 + lr) * SS + koff;
        vf.q[0] = *(const v4u*)vp; vf.q[1] = *(const v4u*)(vp + 8);
        acc3 = __builtin_amdgcn_wmma_f32_16x16x32_bf16(false, pa.bf, false, vf.bf, (short)0, acc3, false, false);
    }

    // ---- normalize + store bf16 row-major (b*S+q, h*64+d)
    for (int r = 0; r < 8; ++r) {
        float l = accl[r];
        float inv = (l > 0.0f) ? 1.0f / l : 0.0f;
        int m = b * SS + q0 + r + half * 8;
        unsigned short* dst = O + (long)m * HIDDEN + h * HD;
        dst[lr]      = f2bf(acc0[r] * inv);
        dst[16 + lr] = f2bf(acc1[r] * inv);
        dst[32 + lr] = f2bf(acc2[r] * inv);
        dst[48 + lr] = f2bf(acc3[r] * inv);
    }
}

// ---------------- host launch ----------------
extern "C" void kernel_launch(void* const* d_in, const int* in_sizes, int n_in,
                              void* d_out, int out_size, void* d_ws, size_t ws_size,
                              hipStream_t stream) {
    (void)in_sizes; (void)n_in; (void)out_size; (void)ws_size;

    const float* X  = (const float*)d_in[0];
    const int* mask = (const int*)d_in[1];
    const float* Wq = (const float*)d_in[2];
    const float* bq = (const float*)d_in[3];
    const float* Wk = (const float*)d_in[4];
    const float* bk = (const float*)d_in[5];
    const float* Wv = (const float*)d_in[6];
    const float* bv = (const float*)d_in[7];
    const float* Wo = (const float*)d_in[8];
    const float* bo = (const float*)d_in[9];
    float* out = (float*)d_out;

    char* ws = (char*)d_ws;
    size_t off = 0;
    auto alloc = [&](size_t bytes) {
        void* p = ws + off;
        off += (bytes + 255) & ~(size_t)255;
        return p;
    };

    const long MR = (long)BB * SS;  // 4096 rows
    unsigned short* Xbf  = (unsigned short*)alloc((size_t)MR * HIDDEN * 2);
    unsigned short* WqT  = (unsigned short*)alloc((size_t)HIDDEN * HIDDEN * 2);
    unsigned short* WkT  = (unsigned short*)alloc((size_t)HIDDEN * KVDIM * 2);
    unsigned short* WvT  = (unsigned short*)alloc((size_t)HIDDEN * KVDIM * 2);
    unsigned short* WoT  = (unsigned short*)alloc((size_t)HIDDEN * HIDDEN * 2);
    unsigned short* Qbf  = (unsigned short*)alloc((size_t)MR * HIDDEN * 2);
    unsigned short* Kbf  = (unsigned short*)alloc((size_t)MR * KVDIM * 2);
    unsigned short* Vtr  = (unsigned short*)alloc((size_t)MR * KVDIM * 2);
    unsigned short* Attn = (unsigned short*)alloc((size_t)MR * HIDDEN * 2);

    // conversions
    {
        long n = MR * HIDDEN;
        cvt_bf16<<<(unsigned)((n + 255) / 256), 256, 0, stream>>>(X, Xbf, n);
        long nw = (long)HIDDEN * HIDDEN;
        transpose_bf16<<<(unsigned)((nw + 255) / 256), 256, 0, stream>>>(Wq, WqT, HIDDEN, HIDDEN);
        transpose_bf16<<<(unsigned)((nw + 255) / 256), 256, 0, stream>>>(Wo, WoT, HIDDEN, HIDDEN);
        long nk = (long)HIDDEN * KVDIM;
        transpose_bf16<<<(unsigned)((nk + 255) / 256), 256, 0, stream>>>(Wk, WkT, HIDDEN, KVDIM);
        transpose_bf16<<<(unsigned)((nk + 255) / 256), 256, 0, stream>>>(Wv, WvT, HIDDEN, KVDIM);
    }

    // projections; Q pre-scaled by (1/sqrt(HD)) * log2(e) so attention runs in exp2 domain
    const float qScale = 0.125f * 1.4426950408889634f;
    {
        int groups = (int)(MR >> 5) * (HIDDEN >> 6);      // 4096
        wmma_gemm_kernel<0><<<(groups + 3) / 4, 128, 0, stream>>>(Xbf, WqT, bq, Qbf, (int)MR, HIDDEN, HIDDEN, qScale);
        int groupsKV = (int)(MR >> 5) * (KVDIM >> 6);     // 1024
        wmma_gemm_kernel<0><<<(groupsKV + 3) / 4, 128, 0, stream>>>(Xbf, WkT, bk, Kbf, (int)MR, KVDIM, HIDDEN, 1.0f);
        wmma_gemm_kernel<1><<<(groupsKV + 3) / 4, 128, 0, stream>>>(Xbf, WvT, bv, Vtr, (int)MR, KVDIM, HIDDEN, 1.0f);
    }

    // attention: B*QH*(S/16) = 8192 waves, 4 waves/block
    gqa_attention_kernel<<<2048, 128, 0, stream>>>(Qbf, Kbf, Vtr, mask, Attn);

    // output projection (fp32 out)
    {
        int groups = (int)(MR >> 5) * (HIDDEN >> 6);
        wmma_gemm_kernel<2><<<(groups + 3) / 4, 128, 0, stream>>>(Attn, WoT, bo, out, (int)MR, HIDDEN, HIDDEN, 1.0f);
    }
}
